// HLMamba_43344809951447
// MI455X (gfx1250) — matrix operand order repaired and checked by reading, not compiled, not run on hardware
//
#include <hip/hip_runtime.h>
#include <cstdint>
#include <cstddef>

typedef _Float16 half_t;
typedef __attribute__((ext_vector_type(16))) _Float16 v16h;
typedef __attribute__((ext_vector_type(8)))  float    v8f;
typedef __attribute__((ext_vector_type(4)))  float    v4f;
typedef __attribute__((ext_vector_type(4)))  unsigned int v4u;
typedef __attribute__((ext_vector_type(4)))  int      v4i;
typedef __attribute__((ext_vector_type(8)))  int      v8i;

#define BN_EPS 1e-5f

#if defined(__has_builtin)
#if __has_builtin(__builtin_amdgcn_tensor_load_to_lds)
#define HAVE_TDM 1
#endif
#endif
#ifndef HAVE_TDM
#define HAVE_TDM 0
#endif

// ---------------------------------------------------------------------------
// f32 -> f16 streaming convert (inputs / K-padded conv weights)
// ---------------------------------------------------------------------------
__global__ void cvt_half_kernel(const float* __restrict__ x, half_t* __restrict__ o, int n) {
  int i = blockIdx.x * blockDim.x + threadIdx.x;
  if (i < n) o[i] = (half_t)x[i];
}

__global__ void wconv_kernel(const float* __restrict__ w, half_t* __restrict__ wh,
                             int Cout, int Ktot, int Kpad) {
  int idx = blockIdx.x * blockDim.x + threadIdx.x;
  if (idx >= Cout * Kpad) return;
  int n = idx / Kpad;
  int k = idx - n * Kpad;
  wh[idx] = (k < Ktot) ? (half_t)w[(size_t)n * Ktot + k] : (half_t)0.0f;
}

// ---------------------------------------------------------------------------
// Implicit-GEMM conv (KHxKW, stride 1) + optional BN + optional ReLU.
// One block per image; input slab staged to LDS. For f16 inputs (INH) the
// staging is a TDM DMA (tensor_load_to_lds, TENSORcnt) when available.
// 4 waves own 16-pixel M-tiles, WMMA f32_16x16x32_f16 over K = Cin*KH*KW.
// A-fragment gather from LDS is branchless (clamped index + select).
// flags: bit0 = BN (gamma/beta), bit1 = ReLU. Bias always applied.
// OUTH: store results as f16 (consumed only by the next conv's WMMA).
// ---------------------------------------------------------------------------
template <int RS, int KWT, int NT, bool INH, bool OUTH>
__global__ void conv_wmma_kernel(const void* __restrict__ xin,
                                 const half_t* __restrict__ wh,
                                 const float* __restrict__ bias,
                                 const float* __restrict__ gamma,
                                 const float* __restrict__ beta,
                                 void* __restrict__ outv,
                                 int Cin, int Cout, int H, int W,
                                 int pad, int Ktot, int Kpad, int flags) {
  extern __shared__ char smemraw[];
  half_t* simg = (half_t*)smemraw;

  const int b   = blockIdx.x;
  const int HWi = H * W;
  const int Nin = Cin * HWi;          // multiple of 4 for all call sites

  if (INH) {
    const half_t* xb = (const half_t*)xin + (size_t)b * Nin;
#if HAVE_TDM
    if (threadIdx.x < 32) {
      unsigned long long ga = (unsigned long long)(uintptr_t)xb;
      unsigned int ldsoff   = (unsigned int)(uintptr_t)simg;
      v4u g0;
      g0[0] = 1u;                                    // count=1, user descriptor
      g0[1] = ldsoff;                                // lds_addr (bytes)
      g0[2] = (unsigned int)(ga & 0xFFFFFFFFu);      // global_addr[31:0]
      g0[3] = (unsigned int)((ga >> 32) & 0x01FFFFFFu) | 0x80000000u; // addr[56:32] | type=2
      v8i g1;
      g1[0] = 0x00010000;                 // data_size = 1 (2 bytes), no mask/pad
      g1[1] = (int)((HWi & 0xFFFF) << 16);           // tensor_dim0[15:0]
      g1[2] = (int)(Cin << 16);                      // dim0 hi | tensor_dim1 lo
      g1[3] = (int)(HWi << 16);                      // dim1 hi | tile_dim0
      g1[4] = Cin;                                   // tile_dim1 | tile_dim2=0
      g1[5] = HWi;                                   // tensor_dim0_stride lo
      g1[6] = 0;                                     // stride0 hi | stride1 lo
      g1[7] = 0;                                     // stride1 hi
      v4i gz;  gz[0] = 0;  gz[1] = 0;  gz[2] = 0;  gz[3] = 0;
#if __clang_major__ >= 23
      v8i gz8; gz8[0] = 0; gz8[1] = 0; gz8[2] = 0; gz8[3] = 0;
      gz8[4] = 0; gz8[5] = 0; gz8[6] = 0; gz8[7] = 0;
      __builtin_amdgcn_tensor_load_to_lds(g0, g1, gz, gz, gz8, 0);
#else
      __builtin_amdgcn_tensor_load_to_lds(g0, g1, gz, gz, 0);
#endif
      __builtin_amdgcn_s_wait_tensorcnt(0);
    }
    __syncthreads();
#else
    const unsigned long long* s64 = (const unsigned long long*)xb;
    int n4 = Nin >> 2;
    for (int i = threadIdx.x; i < n4; i += blockDim.x)
      *(unsigned long long*)(simg + i * 4) = s64[i];
    __syncthreads();
#endif
  } else {
    const float* xb = (const float*)xin + (size_t)b * Nin;
    const int nin4 = Nin >> 2;
    for (int i = threadIdx.x; i < nin4; i += blockDim.x) {
      v4f v = *(const v4f*)(xb + i * 4);
      union { half_t h[4]; unsigned long long u; } t;
      t.h[0] = (half_t)v[0]; t.h[1] = (half_t)v[1];
      t.h[2] = (half_t)v[2]; t.h[3] = (half_t)v[3];
      *(unsigned long long*)(simg + i * 4) = t.u;
    }
    __syncthreads();
  }

  const int lane   = threadIdx.x & 31;
  const int wv     = threadIdx.x >> 5;
  const int nwaves = blockDim.x >> 5;
  const int tiles  = HWi >> 4;         // HW is a multiple of 16 here
  const int lm     = lane & 15;
  const bool hi    = (lane & 16) != 0;
  const int  klo   = hi ? 8 : 0;
  const int  khb   = hi ? 16 : 0;
  const half_t hzero = (half_t)0.0f;

  for (int tile = wv; tile < tiles; tile += nwaves) {
    const int p  = tile * 16 + lm;     // this lane's A-row pixel
    const int py = p / W;
    const int px = p - py * W;

    v8f zero = {};
    v8f acc[NT];
#pragma unroll
    for (int t = 0; t < NT; ++t) acc[t] = zero;

    for (int ks = 0; ks < Kpad; ks += 32) {
      v16h av;
#pragma unroll
      for (int e = 0; e < 16; ++e) {
        int K = ks + khb + e;
        int ci, r, c;
        if (RS == 1) { ci = K; r = 0; c = 0; }
        else {
          ci = K / RS;
          int rem = K - ci * RS;
          r = rem / KWT;
          c = rem - r * KWT;
        }
        int iy = py + r - pad;
        int ix = px + c - pad;
        int ok = (int)(K < Ktot) & (int)((unsigned)iy < (unsigned)H) &
                 (int)((unsigned)ix < (unsigned)W);
        int idx = ok ? (ci * HWi + iy * W + ix) : 0;   // clamped-safe LDS index
        half_t v = simg[idx];                           // unconditional ds load
        av[e] = ok ? v : hzero;                         // value select
      }
#pragma unroll
      for (int nt = 0; nt < NT; ++nt) {
        const int n = nt * 16 + lm;
        v16h bv = *(const v16h*)(wh + (size_t)n * Kpad + ks + khb);
        acc[nt] = __builtin_amdgcn_wmma_f32_16x16x32_f16(
            false, av, false, bv, (short)0, acc[nt], false, false);
      }
    }

#pragma unroll
    for (int nt = 0; nt < NT; ++nt) {
      const int n  = nt * 16 + lm;
      const float bi = bias[n];
      float g = 1.0f, be = 0.0f;
      if (flags & 1) { g = gamma[n] * rsqrtf(1.0f + BN_EPS); be = beta[n]; }
#pragma unroll
      for (int j = 0; j < 8; ++j) {
        int pr = tile * 16 + j + (hi ? 8 : 0);
        float v = acc[nt][j] + bi;
        if (flags & 1) v = v * g + be;
        if (flags & 2) v = fmaxf(v, 0.0f);
        size_t oi = ((size_t)b * Cout + n) * HWi + pr;
        if (OUTH) ((half_t*)outv)[oi] = (half_t)v;
        else      ((float*)outv)[oi]  = v;
      }
    }
  }
}

// ---------------------------------------------------------------------------
// Generic batched GEMM: C[b] = A[b](MxK, lda) * Bw^T (Bw is [N][K], ldb).
// One wave per 16x16 tile. M is a multiple of 16 for all call sites.
// KFULL: K % 32 == 0 -> unguarded b128 gathers; else clamped scalar loads.
// N tail handled by clamping the B row (garbage column never stored).
// FLAGS: bit0 = +bias[N], bit1 = softplus.
// ---------------------------------------------------------------------------
template <bool KFULL, int FLAGS>
__global__ void gemm_wmma_kernel(const float* __restrict__ A,
                                 const float* __restrict__ Bw,
                                 const float* __restrict__ bias,
                                 float* __restrict__ C,
                                 int M, int N, int K,
                                 int lda, int ldb, int ldc,
                                 long long sA, long long sC) {
  const int b  = blockIdx.x;
  const int mt = blockIdx.y;
  const int nt = blockIdx.z;
  const int lane = threadIdx.x & 31;
  const int lm   = lane & 15;
  const bool hi  = (lane & 16) != 0;
  const int klo  = hi ? 8 : 0;
  const int khb  = hi ? 16 : 0;

  const int Mrow = mt * 16 + lm;
  const int Ncol = nt * 16 + lm;
  const bool nv  = Ncol < N;
  const int  Nc  = nv ? Ncol : (N - 1);       // clamped-safe B row

  const float* Arow = A + (size_t)b * sA + (size_t)Mrow * lda;
  const float* Brow = Bw + (size_t)Nc * ldb;
  const half_t hzero = (half_t)0.0f;

  v8f acc = {};
  for (int ks = 0; ks < K; ks += 32) {
    v16h av, bv;
    if (KFULL) {
      const float* ap = Arow + ks + khb;
      v4f a0 = *(const v4f*)(ap);
      v4f a1 = *(const v4f*)(ap + 4);
      v4f a2 = *(const v4f*)(ap + 8);
      v4f a3 = *(const v4f*)(ap + 12);
      const float* bp = Brow + ks + khb;
      v4f b0 = *(const v4f*)(bp);
      v4f b1 = *(const v4f*)(bp + 4);
      v4f b2 = *(const v4f*)(bp + 8);
      v4f b3 = *(const v4f*)(bp + 12);
#pragma unroll
      for (int j = 0; j < 4; ++j) {
        av[j]      = (half_t)a0[j];
        av[4 + j]  = (half_t)a1[j];
        av[8 + j]  = (half_t)a2[j];
        av[12 + j] = (half_t)a3[j];
        bv[j]      = (half_t)b0[j];
        bv[4 + j]  = (half_t)b1[j];
        bv[8 + j]  = (half_t)b2[j];
        bv[12 + j] = (half_t)b3[j];
      }
    } else {
#pragma unroll
      for (int e = 0; e < 16; ++e) {
        int Ka = ks + khb + e;
        int ia = (Ka < K) ? Ka : (K - 1);     // clamped-safe address
        float va = Arow[ia];                  // unconditional load
        av[e] = (Ka < K) ? (half_t)va : hzero;
        int Kb = ks + khb + e;
        int ib = (Kb < K) ? Kb : (K - 1);
        float vb = Brow[ib];
        bv[e] = (Kb < K) ? (half_t)vb : hzero;
      }
    }
    acc = __builtin_amdgcn_wmma_f32_16x16x32_f16(
        false, av, false, bv, (short)0, acc, false, false);
  }

  if (!nv) return;
  const float bi = (FLAGS & 1) ? bias[Ncol] : 0.0f;
#pragma unroll
  for (int j = 0; j < 8; ++j) {
    int r = mt * 16 + j + (hi ? 8 : 0);
    float v = acc[j] + bi;
    if (FLAGS & 2) v = (v > 20.0f) ? v : log1pf(expf(v));
    C[(size_t)b * sC + (size_t)r * ldc + Ncol] = v;
  }
}

// ---------------------------------------------------------------------------
// Elementwise / small kernels
// ---------------------------------------------------------------------------
__global__ void ew_scale_combine_kernel(const float* __restrict__ a,
                                        const float* __restrict__ bb,
                                        const float* __restrict__ cc,
                                        const float* __restrict__ s1p,
                                        const float* __restrict__ s2p,
                                        int order, float* __restrict__ o, int n) {
  int i = blockIdx.x * blockDim.x + threadIdx.x;
  if (i >= n) return;
  float s1 = s1p[0], s2 = s2p[0];
  float t = order ? (bb[i] * s1 + cc[i] * s2) : (cc[i] * s1 + bb[i] * s2);
  o[i] = a[i] * s1 + t * s2;
}

__global__ void ew_trans_pos_kernel(const float* __restrict__ raw,
                                    const float* __restrict__ pos,
                                    float* __restrict__ o) {
  int idx = blockIdx.x * blockDim.x + threadIdx.x;   // b*4096 + t*64 + d
  if (idx >= 256 * 4096) return;
  int b = idx >> 12;
  int r = idx & 4095;
  int t = r >> 6;
  int d = r & 63;
  o[idx] = raw[(size_t)b * 4096 + d * 64 + t] + pos[d];
}

__global__ void ew_mul_kernel(const float* __restrict__ a, float* __restrict__ io, int n) {
  int i = blockIdx.x * blockDim.x + threadIdx.x;
  if (i < n) io[i] = a[i] * io[i];
}

__global__ void ew_add_kernel(const float* __restrict__ a, const float* __restrict__ b,
                              float* __restrict__ o, int n) {
  int i = blockIdx.x * blockDim.x + threadIdx.x;
  if (i < n) o[i] = a[i] + b[i];
}

__global__ void ew_fused_kernel(const float* __restrict__ f1, const float* __restrict__ f2,
                                const float* __restrict__ f3,
                                const float* __restrict__ s1p, const float* __restrict__ s2p,
                                const float* __restrict__ x2, const float* __restrict__ x3,
                                float* __restrict__ o, int n) {
  int i = blockIdx.x * blockDim.x + threadIdx.x;
  if (i >= n) return;
  o[i] = f1[i] + s1p[0] * f2[i] + s2p[0] * f3[i] + x2[i] + x3[i];
}

// Depthwise causal conv (k=4, left pad 3) over L + SiLU. One thread per (b,d).
__global__ void mamba_conv_silu_kernel(const float* __restrict__ xz,
                                       const float* __restrict__ cw,
                                       const float* __restrict__ cb,
                                       float* __restrict__ xm) {
  int idx = blockIdx.x * blockDim.x + threadIdx.x;
  if (idx >= 256 * 128) return;
  int b = idx >> 7;
  int d = idx & 127;
  float w0 = cw[d * 4 + 0], w1 = cw[d * 4 + 1], w2 = cw[d * 4 + 2], w3 = cw[d * 4 + 3];
  float bb = cb[d];
  const float* base = xz + (size_t)b * 64 * 256 + d;   // xm part = first 128 chans
  for (int l = 0; l < 64; ++l) {
    float s = bb;
    if (l >= 3) s += base[(l - 3) * 256] * w0;
    if (l >= 2) s += base[(l - 2) * 256] * w1;
    if (l >= 1) s += base[(l - 1) * 256] * w2;
    s += base[l * 256] * w3;
    xm[((size_t)b * 64 + l) * 128 + d] = s / (1.0f + expf(-s));
  }
}

// Selective-scan core: block = batch, thread = d_inner channel, 16 states/thread.
__global__ void mamba_scan_kernel(const float* __restrict__ xm,
                                  const float* __restrict__ xdbl,
                                  const float* __restrict__ dtb,
                                  const float* __restrict__ xz,
                                  const float* __restrict__ Alog,
                                  const float* __restrict__ Dp,
                                  float* __restrict__ y) {
  int b = blockIdx.x;
  int d = threadIdx.x;   // 128
  float a[16], h[16];
#pragma unroll
  for (int s = 0; s < 16; ++s) { a[s] = -expf(Alog[d * 16 + s]); h[s] = 0.0f; }
  float Dd = Dp[d];
  for (int l = 0; l < 64; ++l) {
    size_t lb = (size_t)b * 64 + l;
    float dt = dtb[lb * 128 + d];
    float xv = xm[lb * 128 + d];
    const float* xd = xdbl + lb * 36;
    float acc = 0.0f;
#pragma unroll
    for (int s = 0; s < 16; ++s) {
      h[s] = expf(dt * a[s]) * h[s] + dt * xd[4 + s] * xv;
      acc += h[s] * xd[20 + s];
    }
    acc += Dd * xv;
    float z = xz[lb * 256 + 128 + d];
    acc *= z / (1.0f + expf(-z));
    y[lb * 128 + d] = acc;
  }
}

// LayerNorm over token 0 of x_vit (64 dims), one wave per batch row.
__global__ void ln_cls_kernel(const float* __restrict__ xvit,
                              const float* __restrict__ g, const float* __restrict__ be,
                              float* __restrict__ o) {
  int b = blockIdx.x;
  int lane = threadIdx.x;
  const float* row = xvit + (size_t)b * 64 * 64;   // token 0
  float v0 = row[lane], v1 = row[lane + 32];
  float s = v0 + v1;
  for (int off = 16; off > 0; off >>= 1) s += __shfl_xor(s, off);
  float mu = s * (1.0f / 64.0f);
  float d0 = v0 - mu, d1 = v1 - mu;
  float q = d0 * d0 + d1 * d1;
  for (int off = 16; off > 0; off >>= 1) q += __shfl_xor(q, off);
  float inv = rsqrtf(q * (1.0f / 64.0f) + 1e-5f);
  o[b * 64 + lane]      = d0 * inv * g[lane]      + be[lane];
  o[b * 64 + lane + 32] = d1 * inv * g[lane + 32] + be[lane + 32];
}

// xm1[b,e] = mean over ch of x1raw[b, ch, e]
__global__ void meanch_kernel(const float* __restrict__ x1raw, float* __restrict__ xm1) {
  int idx = blockIdx.x * blockDim.x + threadIdx.x;
  if (idx >= 256 * 64) return;
  int b = idx >> 6;
  int e = idx & 63;
  float s = 0.0f;
  for (int ch = 0; ch < 64; ++ch) s += x1raw[((size_t)b * 64 + ch) * 64 + e];
  xm1[idx] = s * (1.0f / 64.0f);
}

// out = logits1*coef1 + softmax(logits2)*coef2  (one wave per row; 15 classes)
__global__ void final_combine_kernel(const float* __restrict__ l1,
                                     const float* __restrict__ l2,
                                     const float* __restrict__ c1p,
                                     const float* __restrict__ c2p,
                                     float* __restrict__ o) {
  int b = blockIdx.x;
  int lane = threadIdx.x;
  float v = (lane < 15) ? l2[b * 15 + lane] : -1e30f;
  float m = v;
  for (int off = 16; off > 0; off >>= 1) m = fmaxf(m, __shfl_xor(m, off));
  float e = (lane < 15) ? expf(v - m) : 0.0f;
  float s = e;
  for (int off = 16; off > 0; off >>= 1) s += __shfl_xor(s, off);
  if (lane < 15) o[b * 15 + lane] = l1[b * 15 + lane] * c1p[0] + (e / s) * c2p[0];
}

// ---------------------------------------------------------------------------
// Host orchestration
// ---------------------------------------------------------------------------
extern "C" void kernel_launch(void* const* d_in, const int* in_sizes, int n_in,
                              void* d_out, int out_size, void* d_ws, size_t ws_size,
                              hipStream_t stream) {
  (void)in_sizes; (void)n_in; (void)out_size; (void)ws_size;
  auto P = [&](int i) -> const float* { return (const float*)d_in[i]; };

  const int Bn = 256;
  char* wsb = (char*)d_ws;
  size_t off = 0;
  auto allocF = [&](size_t n) -> float* {
    float* p = (float*)(wsb + off);
    off = (off + n * sizeof(float) + 255) & ~(size_t)255;
    return p;
  };
  auto allocH = [&](size_t n) -> half_t* {
    half_t* p = (half_t*)(wsb + off);
    off = (off + n * sizeof(half_t) + 255) & ~(size_t)255;
    return p;
  };

  // ---- f16 weight + input-image buffers -----------------------------------
  half_t* wh_c1 = allocH((size_t)32 * 1312);
  half_t* wh_c2 = allocH((size_t)32 * 32);
  half_t* wh_s2[6];
  for (int i = 0; i < 6; ++i) wh_s2[i] = allocH((size_t)64 * 288);
  half_t* wh_g1 = allocH((size_t)32 * 32);
  half_t* wh_g2 = allocH((size_t)64 * 32);
  half_t* wh_fm = allocH((size_t)64 * 64);
  const int imgN[9] = {Bn * 144 * 64,  Bn * 64,  Bn * 64,
                       Bn * 144 * 256, Bn * 256, Bn * 256,
                       Bn * 144 * 576, Bn * 576, Bn * 576};
  half_t* imgh[9];
  for (int i = 0; i < 9; ++i) imgh[i] = allocH((size_t)imgN[i]);

  // ---- activation buffers --------------------------------------------------
  const size_t HW3 = 576;
  half_t* hbuf32a = allocH((size_t)Bn * 32 * HW3);   // stage-1 outputs in f16
  half_t* hbuf32b = allocH((size_t)Bn * 32 * HW3);
  half_t* hbuf32c = allocH((size_t)Bn * 32 * HW3);
  float* buf64a = allocF((size_t)Bn * 64 * HW3);
  float* buf64b = allocF((size_t)Bn * 64 * HW3);
  float* buf64c = allocF((size_t)Bn * 64 * HW3);
  float* xadd[3];
  xadd[0] = allocF((size_t)Bn * 64 * 64);
  xadd[1] = allocF((size_t)Bn * 64 * 256);
  xadd[2] = allocF((size_t)Bn * 64 * 576);
  float* x1raw  = allocF((size_t)Bn * 4096);   // raw emb1 output (kept for CNN head)
  float* embtmp = allocF((size_t)Bn * 4096);
  float* xtok[3];
  for (int s = 0; s < 3; ++s) xtok[s] = allocF((size_t)Bn * 4096);
  float* fraw[3];
  for (int s = 0; s < 3; ++s) fraw[s] = allocF((size_t)Bn * 4096);
  float* fb[3];
  for (int s = 0; s < 3; ++s) fb[s] = allocF((size_t)Bn * 4096);
  // mamba temps (reused across branches)
  float* xzb  = allocF((size_t)Bn * 64 * 256);
  float* xmb  = allocF((size_t)Bn * 64 * 128);
  float* xdbl = allocF((size_t)Bn * 64 * 36);
  float* dtb  = allocF((size_t)Bn * 64 * 128);
  float* ybuf = allocF((size_t)Bn * 64 * 128);
  float* mob  = allocF((size_t)Bn * 4096);
  float* preb = allocF((size_t)Bn * 4096);
  float* xvit = allocF((size_t)Bn * 4096);
  float* clsn = allocF((size_t)Bn * 64);
  float* lg1  = allocF((size_t)Bn * 15);
  float* xm1  = allocF((size_t)Bn * 64);
  float* cpool= allocF((size_t)Bn * 32);
  float* lg2  = allocF((size_t)Bn * 15);

  // ---- pre-convert weights (K-padded) and input images to f16 -------------
  auto wc = [&](const float* w, half_t* wh, int Cout, int Ktot, int Kpad) {
    int n = Cout * Kpad;
    wconv_kernel<<<(n + 255) / 256, 256, 0, stream>>>(w, wh, Cout, Ktot, Kpad);
  };
  wc(P(9),  wh_c1, 32, 1296, 1312);
  wc(P(13), wh_c2, 32, 9, 32);
  const int s2w[6] = {17, 21, 25, 29, 33, 37};
  for (int i = 0; i < 6; ++i) wc(P(s2w[i]), wh_s2[i], 64, 288, 288);
  wc(P(41), wh_g1, 32, 1, 32);
  wc(P(43), wh_g2, 64, 32, 32);
  wc(P(54), wh_fm, 64, 64, 64);
  for (int i = 0; i < 9; ++i)
    cvt_half_kernel<<<(imgN[i] + 255) / 256, 256, 0, stream>>>(P(i), imgh[i], imgN[i]);

  auto gemm = [&](const float* A, const float* Bw, const float* bias, float* C,
                  int batch, int M, int N, int K, int lda, int ldb, int ldc,
                  long long sA, long long sC, int flags) {
    dim3 g(batch, (M + 15) / 16, (N + 15) / 16);
    if ((K & 31) == 0) {
      if (flags == 1)
        gemm_wmma_kernel<true, 1><<<g, 32, 0, stream>>>(A, Bw, bias, C, M, N, K,
                                                        lda, ldb, ldc, sA, sC);
      else
        gemm_wmma_kernel<true, 0><<<g, 32, 0, stream>>>(A, Bw, bias, C, M, N, K,
                                                        lda, ldb, ldc, sA, sC);
    } else {
      gemm_wmma_kernel<false, 3><<<g, 32, 0, stream>>>(A, Bw, bias, C, M, N, K,
                                                       lda, ldb, ldc, sA, sC);
    }
  };
  auto ewgrid = [](int n) { return (n + 255) / 256; };

  // ---- multi-scale encoder ------------------------------------------------
  const int Hs[3] = {8, 16, 24};
  const int embW[3] = {47, 49, 51};
  const int embB[3] = {48, 50, 52};
  for (int s = 0; s < 3; ++s) {
    const int H = Hs[s], W = Hs[s], HW = H * W;
    // a = crb(i1; c1)   144->32, 3x3, BN+ReLU  (f16 in, f16 out)
    conv_wmma_kernel<9, 3, 2, true, true><<<Bn, 128, (size_t)144 * HW * 2, stream>>>(
        imgh[0 + s * 3], wh_c1, P(10), P(11), P(12), hbuf32a,
        144, 32, H, W, 1, 1296, 1312, 3);
    // b = crb(i2; c2)   1->32
    conv_wmma_kernel<9, 3, 2, true, true><<<Bn, 128, (size_t)HW * 2, stream>>>(
        imgh[1 + s * 3], wh_c2, P(14), P(15), P(16), hbuf32b,
        1, 32, H, W, 1, 9, 32, 3);
    // c = relu(g1(i3))  1->32, 1x1
    conv_wmma_kernel<1, 1, 2, true, true><<<Bn, 128, (size_t)HW * 2, stream>>>(
        imgh[2 + s * 3], wh_g1, P(42), nullptr, nullptr, hbuf32c,
        1, 32, H, W, 0, 1, 32, 2);
    // stage-2 convs 32->64 (f16 in, f32 out)
    int wi = 17 + s * 8;   // c11/c12/c13
    conv_wmma_kernel<9, 3, 4, true, false><<<Bn, 128, (size_t)32 * HW * 2, stream>>>(
        hbuf32a, wh_s2[2 * s], P(wi + 1), P(wi + 2), P(wi + 3), buf64a,
        32, 64, H, W, 1, 288, 288, 3);
    int wj = 21 + s * 8;   // c21/c22/c23
    conv_wmma_kernel<9, 3, 4, true, false><<<Bn, 128, (size_t)32 * HW * 2, stream>>>(
        hbuf32b, wh_s2[2 * s + 1], P(wj + 1), P(wj + 2), P(wj + 3), buf64b,
        32, 64, H, W, 1, 288, 288, 3);
    conv_wmma_kernel<1, 1, 4, true, false><<<Bn, 128, (size_t)32 * HW * 2, stream>>>(
        hbuf32c, wh_g2, P(44), nullptr, nullptr, buf64c,
        32, 64, H, W, 0, 32, 32, 2);
    // x_add = a*xs1 + (order-dependent mix of b,c)*xs2
    int n = Bn * 64 * HW;
    ew_scale_combine_kernel<<<ewgrid(n), 256, 0, stream>>>(
        buf64a, buf64b, buf64c, P(45), P(46), (s == 2) ? 1 : 0, xadd[s], n);
    // patch embedding GEMM: (b,64,HW) @ embW^T + embB -> (b,64,64)
    float* raw = (s == 0) ? x1raw : embtmp;
    gemm(xadd[s], P(embW[s]), P(embB[s]), raw, Bn, 64, 64, HW, HW, HW, 64,
         (long long)64 * HW, 4096, 1);
    // transpose + pos
    ew_trans_pos_kernel<<<ewgrid(Bn * 4096), 256, 0, stream>>>(raw, P(53), xtok[s]);
  }

  // ---- fm convs (token mixing, no BN/ReLU; f32 in/out) --------------------
  for (int s = 0; s < 3; ++s)
    conv_wmma_kernel<1, 1, 4, false, false><<<Bn, 128, (size_t)64 * 64 * 2, stream>>>(
        xtok[s], wh_fm, P(55), nullptr, nullptr, fraw[s], 64, 64, 8, 8, 0, 64, 64, 0);
  // f2 = f1*f2 ; f3 = f1*f3
  ew_mul_kernel<<<ewgrid(Bn * 4096), 256, 0, stream>>>(fraw[0], fraw[1], Bn * 4096);
  ew_mul_kernel<<<ewgrid(Bn * 4096), 256, 0, stream>>>(fraw[0], fraw[2], Bn * 4096);

  // ---- mamba branches -----------------------------------------------------
  auto run_branch = [&](const float* fin, float* fout) {
    gemm(fin, P(60), nullptr, xzb, Bn, 64, 256, 64, 64, 64, 256, 4096,
         (long long)64 * 256, 0);                                    // in_proj
    mamba_conv_silu_kernel<<<(Bn * 128 + 255) / 256, 256, 0, stream>>>(
        xzb, P(61), P(62), xmb);
    gemm(xmb, P(63), nullptr, xdbl, Bn, 64, 36, 128, 128, 128, 36,
         (long long)64 * 128, (long long)64 * 36, 0);                // x_proj
    gemm(xdbl, P(64), P(65), dtb, Bn, 64, 128, 4, 36, 4, 128,
         (long long)64 * 36, (long long)64 * 128, 3);                // dt + softplus
    mamba_scan_kernel<<<Bn, 128, 0, stream>>>(xmb, xdbl, dtb, xzb, P(66), P(67), ybuf);
    gemm(ybuf, P(68), nullptr, mob, Bn, 64, 64, 128, 128, 128, 64,
         (long long)64 * 128, 4096, 0);                              // out_proj
    ew_add_kernel<<<ewgrid(Bn * 4096), 256, 0, stream>>>(mob, fin, preb, Bn * 4096);
    conv_wmma_kernel<1, 1, 4, false, false><<<Bn, 128, (size_t)64 * 64 * 2, stream>>>(
        preb, wh_fm, P(55), P(56), P(57), fout, 64, 64, 8, 8, 0, 64, 64, 3);
  };
  run_branch(fraw[2], fb[2]);
  run_branch(fraw[1], fb[1]);
  run_branch(fraw[0], fb[0]);

  // ---- fuse + heads -------------------------------------------------------
  ew_fused_kernel<<<ewgrid(Bn * 4096), 256, 0, stream>>>(
      fb[0], fb[1], fb[2], P(58), P(59), xtok[1], xtok[2], xvit, Bn * 4096);
  ln_cls_kernel<<<Bn, 32, 0, stream>>>(xvit, P(69), P(70), clsn);
  gemm(clsn, P(71), P(72), lg1, 1, 256, 15, 64, 64, 64, 15, 0, 0, 1);   // head
  meanch_kernel<<<(Bn * 64 + 255) / 256, 256, 0, stream>>>(x1raw, xm1);
  gemm(xm1, P(73), P(74), cpool, 1, 256, 32, 64, 64, 64, 32, 0, 0, 1);  // cc pool
  gemm(cpool, P(75), P(76), lg2, 1, 256, 15, 32, 32, 32, 15, 0, 0, 1);  // fc
  final_combine_kernel<<<Bn, 32, 0, stream>>>(lg1, lg2, P(77), P(78), (float*)d_out);
}
